// EfficientExpertRouter_85392539779431
// MI455X (gfx1250) — matrix operand order repaired and checked by reading, not compiled, not run on hardware
//
#include <hip/hip_runtime.h>
#include <hip/hip_bf16.h>
#include <math.h>

// ---------------- problem constants ----------------
#define D_MODEL   1024
#define DFF       4096
#define N_EXPERTS 8
#define N_TOK     8192          // 4 * 2048 tokens
#define M_TILE    128           // tokens per FFN block
#define FC        64            // dff chunk per iteration (4 f-tiles)
#define KC        256           // K slab staged per phase-A step
#define NS        256           // N slab staged per phase-B step
#define TILES_PER_EXPERT (N_TOK / M_TILE)   // 64
#define FFN_THREADS 512         // 16 waves -> 4 waves/SIMD, big VGPR budget

typedef __attribute__((ext_vector_type(16))) __bf16 v16bf;
typedef __attribute__((ext_vector_type(2)))  __bf16 v2bf;
typedef __attribute__((ext_vector_type(8)))  float  v8f;

union Frag {
    v16bf    v;
    unsigned u[8];
    uint4    q[2];
};

#if defined(__has_builtin)
#if __has_builtin(__builtin_amdgcn_cvt_pk_bf16_f32)
#define HAVE_CVT_PK_BF16 1
#endif
#endif

// pack two fp32 into packed bf16 pair (1 VALU op)
__device__ __forceinline__ unsigned pack2(float a, float b) {
#ifdef HAVE_CVT_PK_BF16
    v2bf r = __builtin_amdgcn_cvt_pk_bf16_f32(a, b);
    return __builtin_bit_cast(unsigned, r);
#else
    // v_perm_b32: take high halves of a (low result) and b (high result)
    return __builtin_amdgcn_perm(__builtin_bit_cast(unsigned, b),
                                 __builtin_bit_cast(unsigned, a),
                                 0x07060302u);
#endif
}

// A fragment (16x32 bf16) from LDS, row-major, lda_u = row stride in uints.
// Lane l (<16): row l, K = 0..15 ; lane l+16: K = 16..31 (contiguous).
__device__ __forceinline__ void load_a_lds(const unsigned* s, int row_base,
                                           int lda_u, int k0, Frag& a, int lane) {
    int row  = row_base + (lane & 15);
    int koff = (lane >> 4) << 4;                       // 0 or 16 elements
    const uint4* p = reinterpret_cast<const uint4*>(s + row * lda_u + ((k0 + koff) >> 1));
    a.q[0] = p[0];                                     // K = koff .. koff+7
    a.q[1] = p[1];                                     // K = koff+8 .. koff+15
}

// B fragment (32x16 bf16) from LDS where LDS holds the weight slab row-major
// as [n-rows x k-cols]: B[k][n] = slab[row_base + n][k0 + k].
// Lane n (<16): column n, K = 0..15 ; lane n+16: K = 16..31 (contiguous).
__device__ __forceinline__ void load_b_lds(const unsigned* s, int row_base,
                                           int lda_u, int k0, Frag& b, int lane) {
    int row   = row_base + (lane & 15);
    int khalf = (lane >> 4) << 4;                      // 0 or 16 elements
    const uint4* p = reinterpret_cast<const uint4*>(s + row * lda_u + ((k0 + khalf) >> 1));
    b.q[0] = p[0];
    b.q[1] = p[1];
}

__device__ __forceinline__ v8f wmma_bf16(const Frag& a, const Frag& b, v8f c) {
    return __builtin_amdgcn_wmma_f32_16x16x32_bf16(
        false, a.v, false, b.v, (short)0, c, false, false);
}

// ---------------- kernel 0: zero output + expert counters ----------------
__global__ void moe_init_kernel(float4* out4, int* counts, int n4) {
    int i = blockIdx.x * blockDim.x + threadIdx.x;
    if (i < N_EXPERTS) counts[i] = 0;
    float4 z; z.x = 0.f; z.y = 0.f; z.z = 0.f; z.w = 0.f;
    for (; i < n4; i += gridDim.x * blockDim.x) out4[i] = z;
}

// ---------------- kernel 1: router (logits -> softmax -> top2 -> lists) ----
__global__ __launch_bounds__(256)
void moe_router_kernel(const float* __restrict__ x,
                       const float* __restrict__ rw,
                       const float* __restrict__ rb,
                       int* __restrict__ counts,
                       int* __restrict__ toklist,
                       float* __restrict__ wlist) {
    int wave = threadIdx.x >> 5;
    int lane = threadIdx.x & 31;
    int t    = blockIdx.x * 8 + wave;               // one wave per token

    float acc[N_EXPERTS];
#pragma unroll
    for (int e = 0; e < N_EXPERTS; ++e) acc[e] = 0.f;

    const float* xr = x + (size_t)t * D_MODEL;
    for (int d = lane; d < D_MODEL; d += 32) {
        float xv = xr[d];
        const float4* w = reinterpret_cast<const float4*>(rw + d * N_EXPERTS);
        float4 w0 = w[0], w1 = w[1];
        acc[0] += xv * w0.x; acc[1] += xv * w0.y;
        acc[2] += xv * w0.z; acc[3] += xv * w0.w;
        acc[4] += xv * w1.x; acc[5] += xv * w1.y;
        acc[6] += xv * w1.z; acc[7] += xv * w1.w;
    }
#pragma unroll
    for (int e = 0; e < N_EXPERTS; ++e)
#pragma unroll
        for (int off = 16; off > 0; off >>= 1)
            acc[e] += __shfl_xor(acc[e], off, 32);

    if (lane == 0) {
        float l[N_EXPERTS], m = -1e30f;
#pragma unroll
        for (int e = 0; e < N_EXPERTS; ++e) { l[e] = acc[e] + rb[e]; m = fmaxf(m, l[e]); }
        float s = 0.f;
#pragma unroll
        for (int e = 0; e < N_EXPERTS; ++e) { l[e] = __expf(l[e] - m); s += l[e]; }
        float inv = 1.f / s;
        int   i1 = 0; float p1 = -1.f;
#pragma unroll
        for (int e = 0; e < N_EXPERTS; ++e) { float p = l[e] * inv; if (p > p1) { p1 = p; i1 = e; } }
        int   i2 = -1; float p2 = -1.f;
#pragma unroll
        for (int e = 0; e < N_EXPERTS; ++e) {
            float p = l[e] * inv;
            if (e != i1 && p > p2) { p2 = p; i2 = e; }
        }
        int s1 = atomicAdd(&counts[i1], 1);
        toklist[i1 * N_TOK + s1] = t;  wlist[i1 * N_TOK + s1] = p1;
        int s2 = atomicAdd(&counts[i2], 1);
        toklist[i2 * N_TOK + s2] = t;  wlist[i2 * N_TOK + s2] = p2;
    }
}

// ---------------- kernel 2: fused expert FFN (up -> gelu -> down) ----------
// Block: 512 threads = 16 waves, one (expert, 128-token tile).
// x tile LDS-resident (bf16). Weights staged cooperatively through a shared
// 32KB LDS slab (each weight element converted to bf16 ONCE per block).
// Phase A: h[128xFC] = gelu(x @ up_chunk^T), K slabbed by KC.
//          wave (mi2 = w&7, wh = w>>3): h tiles (mi2, f-tiles wh*2+{0,1});
//          one A fragment feeds both WMMAs.
// Phase B: out[128x1024] += h @ down_chunk^T, N slabbed by NS.
//          wave owns m-tile mi2 and 32 n-tiles (8 per slab) -> acc[32].
__global__ __launch_bounds__(FFN_THREADS)
__attribute__((amdgpu_waves_per_eu(4, 4)))
void moe_ffn_kernel(const float* __restrict__ x,
                    const float* __restrict__ up,
                    const float* __restrict__ down,
                    const int*  __restrict__ counts,
                    const int*  __restrict__ toklist,
                    const float* __restrict__ wlist,
                    float* __restrict__ out) {
    const int e    = blockIdx.x / TILES_PER_EXPERT;
    const int tile = blockIdx.x % TILES_PER_EXPERT;
    const int cnt  = counts[e];
    const int base = tile * M_TILE;
    if (base >= cnt) return;

    const int tid  = threadIdx.x;
    const int wave = tid >> 5;
    const int lane = tid & 31;
    const int mi2  = wave & 7;                        // m-tile 0..7
    const int wh   = wave >> 3;                       // 0..1 (f-pair / n-group)

    __shared__ alignas(16) unsigned xs[M_TILE * (D_MODEL / 2)];   // 128x1024 bf16 = 256 KB
    __shared__ alignas(16) __bf16   hs[M_TILE * FC];              // 128x64  bf16 = 16 KB
    __shared__ alignas(16) unsigned wslab[(FC * KC) / 2];         // 32 KB, reused A/B
    __shared__ int   toks[M_TILE];
    __shared__ float tws[M_TILE];

    if (tid < M_TILE) {
        int s = base + tid;
        if (s < cnt) { toks[tid] = toklist[e * N_TOK + s]; tws[tid] = wlist[e * N_TOK + s]; }
        else         { toks[tid] = 0;                      tws[tid] = 0.f; }
    }
    __syncthreads();

    // stage gathered x rows into LDS as bf16 (read from HBM exactly once)
    for (int i = tid; i < M_TILE * (D_MODEL / 4); i += FFN_THREADS) {
        int row = i >> 8;                              // 256 float4 per row
        int c4  = i & 255;
        float4 f = *reinterpret_cast<const float4*>(x + (size_t)toks[row] * D_MODEL + c4 * 4);
        xs[row * (D_MODEL / 2) + c4 * 2]     = pack2(f.x, f.y);
        xs[row * (D_MODEL / 2) + c4 * 2 + 1] = pack2(f.z, f.w);
    }

    const float* upE = up   + (size_t)e * DFF * D_MODEL;   // [DFF x D] row-major
    const float* dnE = down + (size_t)e * D_MODEL * DFF;   // [D x DFF] row-major
    const unsigned* hsu = reinterpret_cast<const unsigned*>(hs);

    v8f acc[32];                                      // out: m-tile mi2, 32 n-tiles
#pragma unroll
    for (int i = 0; i < 32; ++i)
#pragma unroll
        for (int r = 0; r < 8; ++r) acc[i][r] = 0.f;

    for (int ch = 0; ch < DFF / FC; ++ch) {
        const int fr0 = ch * FC;

        // ======== phase A: h = x[128x1024] @ up[fr0..fr0+64][1024]^T ========
        v8f hacc[2];
#pragma unroll
        for (int t = 0; t < 2; ++t)
#pragma unroll
            for (int r = 0; r < 8; ++r) hacc[t][r] = 0.f;

#pragma unroll 1
        for (int kc = 0; kc < D_MODEL; kc += KC) {
            __syncthreads();                           // wslab free to overwrite
            // stage up slab [FC x KC] as bf16 (row stride KC/2 uints)
            for (int i = tid; i < (FC * KC) / 4; i += FFN_THREADS) {
                int r  = i >> 6;                       // KC/4 = 64 float4 per row
                int c4 = i & 63;
                float4 f = *reinterpret_cast<const float4*>(
                    upE + (size_t)(fr0 + r) * D_MODEL + kc + c4 * 4);
                wslab[r * (KC / 2) + c4 * 2]     = pack2(f.x, f.y);
                wslab[r * (KC / 2) + c4 * 2 + 1] = pack2(f.z, f.w);
            }
            __syncthreads();

#pragma unroll 1
            for (int k0 = 0; k0 < KC; k0 += 32) {
                Frag a, b0, b1;
                load_a_lds(xs, mi2 * 16, D_MODEL / 2, kc + k0, a, lane);
                load_b_lds(wslab, (wh * 2 + 0) * 16, KC / 2, k0, b0, lane);
                hacc[0] = wmma_bf16(a, b0, hacc[0]);
                load_b_lds(wslab, (wh * 2 + 1) * 16, KC / 2, k0, b1, lane);
                hacc[1] = wmma_bf16(a, b1, hacc[1]);
            }
        }

        // gelu + store h tiles to LDS
#pragma unroll
        for (int t = 0; t < 2; ++t)
#pragma unroll
            for (int r = 0; r < 8; ++r) {
                float v = hacc[t][r];
                v = 0.5f * v * (1.f + erff(v * 0.70710678118654752f));
                int m  = mi2 * 16 + r + ((lane >> 4) << 3);
                int fl = (wh * 2 + t) * 16 + (lane & 15);
                hs[m * FC + fl] = (__bf16)v;
            }

        // ======== phase B: out += h[128x64] @ down[:, fr0..+64]^T ========
#pragma unroll
        for (int ns = 0; ns < D_MODEL / NS; ++ns) {
            __syncthreads();                           // hs visible, wslab free
            // stage down slab [NS x FC] as bf16 (row stride FC/2 uints)
            const int n0s = ns * NS;
            for (int i = tid; i < (NS * FC) / 4; i += FFN_THREADS) {
                int r  = i >> 4;                       // FC/4 = 16 float4 per row
                int c4 = i & 15;
                float4 f = *reinterpret_cast<const float4*>(
                    dnE + (size_t)(n0s + r) * DFF + fr0 + c4 * 4);
                wslab[r * (FC / 2) + c4 * 2]     = pack2(f.x, f.y);
                wslab[r * (FC / 2) + c4 * 2 + 1] = pack2(f.z, f.w);
            }
            __syncthreads();

            // h A-fragments for this wave's m-tile (reloaded per slab to
            // keep live ranges short)
            Frag ha0, ha1;
            load_a_lds(hsu, mi2 * 16, FC / 2, 0,  ha0, lane);
            load_a_lds(hsu, mi2 * 16, FC / 2, 32, ha1, lane);

#pragma unroll
            for (int j = 0; j < 8; ++j) {
                const int ntl = wh * 8 + j;            // n-tile within slab
                const int ai  = ns * 8 + j;
                Frag b;
                load_b_lds(wslab, ntl * 16, FC / 2, 0, b, lane);
                acc[ai] = wmma_bf16(ha0, b, acc[ai]);
                load_b_lds(wslab, ntl * 16, FC / 2, 32, b, lane);
                acc[ai] = wmma_bf16(ha1, b, acc[ai]);
            }
        }
    }

    // ---- writeback: out[tok, :] += w * acc (each element gets exactly 2 adds) ----
#pragma unroll
    for (int ns = 0; ns < 4; ++ns)
#pragma unroll
        for (int j = 0; j < 8; ++j) {
            int ai  = ns * 8 + j;
            int col = ns * NS + (wh * 8 + j) * 16 + (lane & 15);
#pragma unroll
            for (int r = 0; r < 8; ++r) {
                int   m = mi2 * 16 + r + ((lane >> 4) << 3);
                float w = tws[m];
                if (w != 0.f) {
                    unsafeAtomicAdd(out + (size_t)toks[m] * D_MODEL + col, acc[ai][r] * w);
                }
            }
        }
}

// ---------------- host launcher ----------------
extern "C" void kernel_launch(void* const* d_in, const int* in_sizes, int n_in,
                              void* d_out, int out_size, void* d_ws, size_t ws_size,
                              hipStream_t stream) {
    (void)in_sizes; (void)n_in; (void)out_size; (void)ws_size;
    const float* x    = (const float*)d_in[0];
    const float* rw   = (const float*)d_in[1];
    const float* rb   = (const float*)d_in[2];
    const float* up   = (const float*)d_in[3];
    const float* down = (const float*)d_in[4];
    float* out = (float*)d_out;

    char*  ws      = (char*)d_ws;
    int*   counts  = (int*)ws;                                   // 8 ints
    int*   toklist = (int*)(ws + 256);                           // 8*8192 ints
    float* wlist   = (float*)(ws + 256 + N_EXPERTS * N_TOK * 4); // 8*8192 floats

    const int n4 = N_TOK * D_MODEL / 4;
    moe_init_kernel<<<2048, 256, 0, stream>>>((float4*)out, counts, n4);

    moe_router_kernel<<<N_TOK / 8, 256, 0, stream>>>(x, rw, rb, counts, toklist, wlist);

    moe_ffn_kernel<<<N_EXPERTS * TILES_PER_EXPERT, FFN_THREADS, 0, stream>>>(
        x, up, down, counts, toklist, wlist, out);
}
